// CombinedGoalObsNetwork_40948218200135
// MI455X (gfx1250) — matrix operand order repaired and checked by reading, not compiled, not run on hardware
//
#include <hip/hip_runtime.h>
#include <hip/hip_bf16.h>

#define F 64

typedef __attribute__((ext_vector_type(16))) _Float16 v16h;
typedef __attribute__((ext_vector_type(8)))  float    v8f;

// ---------------------------------------------------------------------------
// Guaranteed single-instruction f32 atomic add (no CAS loop), device scope so
// the read-modify-write is performed coherently at L2 (agg buffers are 8 MB,
// fully L2-resident on the 192 MB L2).
// ---------------------------------------------------------------------------
__device__ __forceinline__ void atomic_add_f32(float* p, float v) {
    asm volatile("global_atomic_add_f32 %0, %1, off scope:SCOPE_DEV"
                 :: "v"(p), "v"(v) : "memory");
}

// ---------------------------------------------------------------------------
// Zero the accumulation buffers (agg + agg2), float4-wide.
// ---------------------------------------------------------------------------
__global__ void k_zero(float4* p, long n4) {
    long i = (long)blockIdx.x * blockDim.x + threadIdx.x;
    if (i < n4) p[i] = make_float4(0.f, 0.f, 0.f, 0.f);
}

// ---------------------------------------------------------------------------
// Pre-pack a 64x64 f32 weight matrix (k-major) into f16 laid out exactly as
// the WMMA B operand registers: [kstep(2)][ntile(4)][lane(32)][e(16)].
// B layout (ISA 7.12.2, 16-bit B 32x16): lane L -> col L&15, elem e -> K =
// (e&7) + 8*(L>=16) + 16*(e>=8). One lane tile = 16 contiguous halfs = 32 B.
// ---------------------------------------------------------------------------
__global__ void k_pack(const float* __restrict__ W, _Float16* __restrict__ Wp) {
    int t = blockIdx.x * blockDim.x + threadIdx.x;   // 4096 threads
    int e  = t & 15;
    int ln = (t >> 4) & 31;
    int nt = (t >> 9) & 3;
    int kb = (t >> 11) & 1;
    int col = ln & 15, hi = (ln >> 4) & 1;
    int k = (e & 7) + hi * 8 + (e >> 3) * 16;
    Wp[t] = (_Float16)W[(kb * 32 + k) * F + nt * 16 + col];
}

// ---------------------------------------------------------------------------
// Edge phase 1: msg = relu(x_state[src] + edge_attr @ We + be); atomic-add
// into agg[dst]. 16 threads per edge, float4 per thread.
// ---------------------------------------------------------------------------
__global__ void k_edge1(const float* __restrict__ xs, const float* __restrict__ ea,
                        const float* __restrict__ We, const float* __restrict__ be,
                        const int* __restrict__ src, const int* __restrict__ dst,
                        float* __restrict__ agg, int E) {
    long t = (long)blockIdx.x * blockDim.x + threadIdx.x;
    int e = (int)(t >> 4), q = (int)(t & 15);
    if (e >= E) return;
    int s = src[e], d = dst[e];
    float a0 = ea[(long)e * 2 + 0];
    float a1 = ea[(long)e * 2 + 1];
    float4 w0 = *(const float4*)(We + q * 4);
    float4 w1 = *(const float4*)(We + F + q * 4);
    float4 bb = *(const float4*)(be + q * 4);
    float4 xv = *(const float4*)(xs + (long)s * F + q * 4);
    float4 m;
    m.x = fmaxf(xv.x + a0 * w0.x + a1 * w1.x + bb.x, 0.f);
    m.y = fmaxf(xv.y + a0 * w0.y + a1 * w1.y + bb.y, 0.f);
    m.z = fmaxf(xv.z + a0 * w0.z + a1 * w1.z + bb.z, 0.f);
    m.w = fmaxf(xv.w + a0 * w0.w + a1 * w1.w + bb.w, 0.f);
    float* p = agg + (long)d * F + q * 4;
    atomic_add_f32(p + 0, m.x);
    atomic_add_f32(p + 1, m.y);
    atomic_add_f32(p + 2, m.z);
    atomic_add_f32(p + 3, m.w);
}

// ---------------------------------------------------------------------------
// Edge phase 2: atomic scatter of x1[src] into agg2[dst].
// ---------------------------------------------------------------------------
__global__ void k_edge2(const float* __restrict__ x1,
                        const int* __restrict__ src, const int* __restrict__ dst,
                        float* __restrict__ agg2, int E) {
    long t = (long)blockIdx.x * blockDim.x + threadIdx.x;
    int e = (int)(t >> 4), q = (int)(t & 15);
    if (e >= E) return;
    int s = src[e], d = dst[e];
    float4 xv = *(const float4*)(x1 + (long)s * F + q * 4);
    float* p = agg2 + (long)d * F + q * 4;
    atomic_add_f32(p + 0, xv.x);
    atomic_add_f32(p + 1, xv.y);
    atomic_add_f32(p + 2, xv.z);
    atomic_add_f32(p + 3, xv.w);
}

// ---------------------------------------------------------------------------
// WMMA A operand from f16 LDS tile (16xF row-major). 16-bit A 16x32 layout
// collapses to: a[e] = hs[row][kb + e + 8*hi] (e<8), hs[row][kb+16+(e-8)+8*hi]
// (e>=8) -> two contiguous 16-byte chunks per lane (2x ds_load_b128).
// ---------------------------------------------------------------------------
__device__ __forceinline__ v16h load_a16(const _Float16* hs, int lane, int kb) {
    int row = lane & 15, hi = (lane >> 4) & 1;
    const _Float16* r = hs + row * F + kb + hi * 8;
    v16h a;
#pragma unroll
    for (int e = 0; e < 8; ++e) a[e] = r[e];
#pragma unroll
    for (int e = 0; e < 8; ++e) a[8 + e] = r[16 + e];
    return a;
}

// ---------------------------------------------------------------------------
// MLP1: h = x_task + agg; x1 = relu(h@W1a + b1a)@W1b + b1b
// One block = 16 rows, 4 waves; each wave owns a 16-col N-tile.
// B operands come pre-packed (one contiguous 32 B load per lane per k-step).
// ---------------------------------------------------------------------------
__global__ __launch_bounds__(128) void k_mlp1(
    const float* __restrict__ x_task, const float* __restrict__ agg,
    const v16h* __restrict__ Bp1a, const float* __restrict__ b1a,
    const v16h* __restrict__ Bp1b, const float* __restrict__ b1b,
    float* __restrict__ x1) {
    __shared__ _Float16 hs[16 * F];
    __shared__ _Float16 ts[16 * F];
    int tid = threadIdx.x;
    long base = (long)blockIdx.x * 16 * F;
    for (int i = tid; i < 16 * F; i += 128)
        hs[i] = (_Float16)(x_task[base + i] + agg[base + i]);
    __syncthreads();

    int lane = tid & 31, wave = tid >> 5;
    int ntile = wave * 16, col = lane & 15, hi = (lane >> 4) & 1;

    v8f c;
#pragma unroll
    for (int v = 0; v < 8; ++v) c[v] = b1a[ntile + col];
#pragma unroll
    for (int ks = 0; ks < 2; ++ks) {
        v16h a = load_a16(hs, lane, ks * 32);
        v16h b = Bp1a[(ks * 4 + wave) * 32 + lane];
        c = __builtin_amdgcn_wmma_f32_16x16x32_f16(false, a, false, b, (short)0, c,
                                                   false, false);
    }
#pragma unroll
    for (int v = 0; v < 8; ++v)
        ts[(v + hi * 8) * F + ntile + col] = (_Float16)fmaxf(c[v], 0.f);
    __syncthreads();

#pragma unroll
    for (int v = 0; v < 8; ++v) c[v] = b1b[ntile + col];
#pragma unroll
    for (int ks = 0; ks < 2; ++ks) {
        v16h a = load_a16(ts, lane, ks * 32);
        v16h b = Bp1b[(ks * 4 + wave) * 32 + lane];
        c = __builtin_amdgcn_wmma_f32_16x16x32_f16(false, a, false, b, (short)0, c,
                                                   false, false);
    }
#pragma unroll
    for (int v = 0; v < 8; ++v) x1[base + (v + hi * 8) * F + ntile + col] = c[v];
}

// ---------------------------------------------------------------------------
// MLP2: h2 = x_actor + agg2; t = relu(h2@W2a + b2a); logits = t@W2b + b2b
// WMMA first layer, VALU matvec head (N=1).
// ---------------------------------------------------------------------------
__global__ __launch_bounds__(128) void k_mlp2(
    const float* __restrict__ x_actor, const float* __restrict__ agg2,
    const v16h* __restrict__ Bp2a, const float* __restrict__ b2a,
    const float* __restrict__ W2b, const float* __restrict__ b2b,
    float* __restrict__ out) {
    __shared__ _Float16 hs[16 * F];
    __shared__ _Float16 ts[16 * F];
    int tid = threadIdx.x;
    long base = (long)blockIdx.x * 16 * F;
    for (int i = tid; i < 16 * F; i += 128)
        hs[i] = (_Float16)(x_actor[base + i] + agg2[base + i]);
    __syncthreads();

    int lane = tid & 31, wave = tid >> 5;
    int ntile = wave * 16, col = lane & 15, hi = (lane >> 4) & 1;

    v8f c;
#pragma unroll
    for (int v = 0; v < 8; ++v) c[v] = b2a[ntile + col];
#pragma unroll
    for (int ks = 0; ks < 2; ++ks) {
        v16h a = load_a16(hs, lane, ks * 32);
        v16h b = Bp2a[(ks * 4 + wave) * 32 + lane];
        c = __builtin_amdgcn_wmma_f32_16x16x32_f16(false, a, false, b, (short)0, c,
                                                   false, false);
    }
#pragma unroll
    for (int v = 0; v < 8; ++v)
        ts[(v + hi * 8) * F + ntile + col] = (_Float16)fmaxf(c[v], 0.f);
    __syncthreads();

    if (tid < 16) {
        float s = b2b[0];
#pragma unroll
        for (int cc = 0; cc < F; ++cc) s += (float)ts[tid * F + cc] * W2b[cc];
        out[(long)blockIdx.x * 16 + tid] = s;
    }
}

// ---------------------------------------------------------------------------
extern "C" void kernel_launch(void* const* d_in, const int* in_sizes, int n_in,
                              void* d_out, int out_size, void* d_ws, size_t ws_size,
                              hipStream_t stream) {
    const float* x_state   = (const float*)d_in[0];
    const float* x_task    = (const float*)d_in[1];
    const float* x_actor   = (const float*)d_in[2];
    const float* edge_attr = (const float*)d_in[3];
    const float* We  = (const float*)d_in[4];
    const float* be  = (const float*)d_in[5];
    const float* W1a = (const float*)d_in[6];
    const float* b1a = (const float*)d_in[7];
    const float* W1b = (const float*)d_in[8];
    const float* b1b = (const float*)d_in[9];
    const float* W2a = (const float*)d_in[10];
    const float* b2a = (const float*)d_in[11];
    const float* W2b = (const float*)d_in[12];
    const float* b2b = (const float*)d_in[13];
    const int* src_st = (const int*)d_in[14];
    const int* dst_st = (const int*)d_in[15];
    const int* src_ta = (const int*)d_in[16];
    const int* dst_ta = (const int*)d_in[17];

    int n_task  = in_sizes[1] / F;
    int n_actor = in_sizes[2] / F;
    int E_st = in_sizes[14];
    int E_ta = in_sizes[16];

    float* agg  = (float*)d_ws;                       // [n_task, F]
    float* agg2 = agg  + (long)n_task * F;            // [n_actor, F]
    float* x1   = agg2 + (long)n_actor * F;           // [n_task, F]
    _Float16* Wp1a = (_Float16*)(x1 + (long)n_task * F);
    _Float16* Wp1b = Wp1a + 4096;
    _Float16* Wp2a = Wp1b + 4096;

    // 1) zero accumulators (agg + agg2 are contiguous)
    long n4 = ((long)(n_task + n_actor) * F) / 4;
    k_zero<<<(unsigned)((n4 + 255) / 256), 256, 0, stream>>>((float4*)agg, n4);

    // 2) pre-pack GEMM B operands into WMMA register layout (f16)
    k_pack<<<16, 256, 0, stream>>>(W1a, Wp1a);
    k_pack<<<16, 256, 0, stream>>>(W1b, Wp1b);
    k_pack<<<16, 256, 0, stream>>>(W2a, Wp2a);

    // 3) GINEConv edge scatter
    long t1 = (long)E_st * 16;
    k_edge1<<<(unsigned)((t1 + 255) / 256), 256, 0, stream>>>(
        x_state, edge_attr, We, be, src_st, dst_st, agg, E_st);

    // 4) task MLP (WMMA)
    k_mlp1<<<n_task / 16, 128, 0, stream>>>(x_task, agg, (const v16h*)Wp1a, b1a,
                                            (const v16h*)Wp1b, b1b, x1);

    // 5) GINConv edge scatter
    long t2 = (long)E_ta * 16;
    k_edge2<<<(unsigned)((t2 + 255) / 256), 256, 0, stream>>>(
        x1, src_ta, dst_ta, agg2, E_ta);

    // 6) actor MLP + scoring head (WMMA + matvec)
    k_mlp2<<<n_actor / 16, 128, 0, stream>>>(x_actor, agg2, (const v16h*)Wp2a, b2a,
                                             W2b, b2b, (float*)d_out);
}
